// SNNBackend_75196287418806
// MI455X (gfx1250) — matrix operand order repaired and checked by reading, not compiled, not run on hardware
//
#include <hip/hip_runtime.h>
#include <hip/hip_bf16.h>

// ---------------------------------------------------------------------------
// Fused SNN forward for MI455X (gfx1250):
//   current = x @ W^T + b  (bf16 WMMA, fp32 accum), fused with the LIF scan.
// Grid: (H/128, B) workgroups of 128 threads (4 waves, wave32).
// Each WG: caches its 128x512 W tile in LDS (bf16), then loops over T in
// chunks of 16 timesteps: stage x chunk (bf16) -> v_wmma_f32_16x16x32_bf16
// -> stage 16x128 current tile -> per-thread LIF scan (v,i in registers).
// Avoids the 537 MB round-trip of the `current` intermediate entirely.
// ---------------------------------------------------------------------------

typedef __attribute__((ext_vector_type(16))) __bf16 v16bf;
typedef __attribute__((ext_vector_type(8)))  float  v8f;

#define B_DIM   32
#define T_DIM   2048
#define I_DIM   512
#define H_DIM   1024

#define TPB     128          // 4 waves
#define H_TILE  128
#define T_CHUNK 16
#define LDS_STRIDE (I_DIM + 8)    // halves; +8 breaks 64-bank alignment
#define CUR_STRIDE (H_TILE + 4)   // floats

// LIF constants (DT=1): v_dec = v + 0.1*((0 - v) + i); i_dec = 0.8*i
#define TAU_MEM_INV 0.1f
#define TAU_SYN_INV 0.2f
#define V_TH        1.0f

__device__ __forceinline__ unsigned short f32_to_bf16_rne(float f) {
    unsigned int u = __float_as_uint(f);
    u += 0x7FFFu + ((u >> 16) & 1u);
    return (unsigned short)(u >> 16);
}

__global__ __launch_bounds__(TPB)
void snn_fused_kernel(const float* __restrict__ x,
                      const float* __restrict__ W,
                      const float* __restrict__ bias,
                      float* __restrict__ out)
{
    extern __shared__ char smem_raw[];
    unsigned short* Wlds = (unsigned short*)smem_raw;                 // [H_TILE][LDS_STRIDE]
    unsigned short* xs   = Wlds + (size_t)H_TILE * LDS_STRIDE;        // [T_CHUNK][LDS_STRIDE]
    float*          cur  = (float*)(xs + (size_t)T_CHUNK * LDS_STRIDE); // [T_CHUNK][CUR_STRIDE]

    const int tid = threadIdx.x;
    const int b   = blockIdx.y;
    const int h0  = blockIdx.x * H_TILE;

    // ---- load W tile -> bf16 LDS (coalesced; W is 2MB, lives in L2) ----
    for (int idx = tid; idx < H_TILE * I_DIM; idx += TPB) {
        const int r = idx >> 9;            // /512
        const int c = idx & (I_DIM - 1);
        Wlds[r * LDS_STRIDE + c] = f32_to_bf16_rne(W[(size_t)(h0 + r) * I_DIM + c]);
    }

    const int lane = tid & 31;
    const int wv   = tid >> 5;     // wave 0..3
    const int mrow = lane & 15;    // row/col index inside 16x16 tile
    const int hi   = lane >> 4;    // half-wave select

    float v_mem = 0.0f, i_syn = 0.0f;
    const float bval = bias[h0 + tid];

    __syncthreads();

    const float* xb     = x   + (size_t)b * T_DIM * I_DIM;
    float*       spikes = out + (size_t)b * T_DIM * H_DIM;

    // Fragment base pointers into LDS (contiguous k order, same for A and B):
    // A (16x32, bf16): lane m<16 -> row m, K k0..k0+15 contiguous;
    //                  lanes 16-31 -> K k0+16..k0+31.  Two b128 LDS loads.
    // B (32x16, bf16): lane n<16 -> col n (= W row), K k0..k0+15 contiguous;
    //                  lanes 16-31 -> K k0+16..k0+31.  Two b128 LDS loads.
    const unsigned short* arow  = &xs[mrow * LDS_STRIDE + hi * 16];
    const unsigned short* brow0 = &Wlds[(wv * 32 +      mrow) * LDS_STRIDE + hi * 16];
    const unsigned short* brow1 = &Wlds[(wv * 32 + 16 + mrow) * LDS_STRIDE + hi * 16];

    for (int t0 = 0; t0 < T_DIM; t0 += T_CHUNK) {
        // ---- stage x chunk -> bf16 LDS ----
        const float* xc = xb + (size_t)t0 * I_DIM;
        for (int idx = tid; idx < T_CHUNK * I_DIM; idx += TPB) {
            const int r = idx >> 9;
            const int c = idx & (I_DIM - 1);
            xs[r * LDS_STRIDE + c] = f32_to_bf16_rne(xc[(size_t)r * I_DIM + c]);
        }
        // prefetch next chunk's x (gfx1250 global_prefetch_b8)
        if (t0 + T_CHUNK < T_DIM)
            __builtin_prefetch(xc + (size_t)T_CHUNK * I_DIM + tid * 16, 0, 1);
        __syncthreads();

        // ---- chunk GEMM: 16(t) x 128(h), K=512 in 16 steps of 32 ----
        v8f acc0 = {};
        v8f acc1 = {};
        #pragma unroll
        for (int k0 = 0; k0 < I_DIM; k0 += 32) {
            v16bf a, b0, b1;
            ((uint4*)&a )[0] = *(const uint4*)(arow  + k0);
            ((uint4*)&a )[1] = *(const uint4*)(arow  + k0 + 8);
            ((uint4*)&b0)[0] = *(const uint4*)(brow0 + k0);
            ((uint4*)&b0)[1] = *(const uint4*)(brow0 + k0 + 8);
            ((uint4*)&b1)[0] = *(const uint4*)(brow1 + k0);
            ((uint4*)&b1)[1] = *(const uint4*)(brow1 + k0 + 8);
            acc0 = __builtin_amdgcn_wmma_f32_16x16x32_bf16(
                       false, a, false, b0, (short)0, acc0, false, false);
            acc1 = __builtin_amdgcn_wmma_f32_16x16x32_bf16(
                       false, a, false, b1, (short)0, acc1, false, false);
        }

        // ---- stage current tile to LDS (C layout: VGPR r -> t=r+hi*8, lane n -> h) ----
        {
            const int hl0 = wv * 32 + mrow;
            const int hl1 = hl0 + 16;
            #pragma unroll
            for (int r = 0; r < 8; ++r) {
                const int t = r + hi * 8;
                cur[t * CUR_STRIDE + hl0] = acc0[r];
                cur[t * CUR_STRIDE + hl1] = acc1[r];
            }
        }
        __syncthreads();

        // ---- LIF scan: one h per thread, strict t order, v/i in registers ----
        float* srow = spikes + (size_t)t0 * H_DIM + h0 + tid;
        #pragma unroll
        for (int t = 0; t < T_CHUNK; ++t) {
            const float c_t   = cur[t * CUR_STRIDE + tid] + bval;
            const float v_dec = v_mem + TAU_MEM_INV * ((0.0f - v_mem) + i_syn);
            const float i_dec = i_syn - TAU_SYN_INV * i_syn;
            const float z     = (v_dec - V_TH > 0.0f) ? 1.0f : 0.0f;
            v_mem = (1.0f - z) * v_dec;   // + z * V_RESET (= 0)
            i_syn = i_dec + c_t;
            srow[(size_t)t * H_DIM] = z;
        }
        __syncthreads();
    }

    // ---- final state: v_f then i_f, appended after spikes ----
    const size_t base = (size_t)B_DIM * T_DIM * H_DIM;
    out[base + (size_t)b * H_DIM + h0 + tid]                          = v_mem;
    out[base + (size_t)B_DIM * H_DIM + (size_t)b * H_DIM + h0 + tid]  = i_syn;
}

extern "C" void kernel_launch(void* const* d_in, const int* in_sizes, int n_in,
                              void* d_out, int out_size, void* d_ws, size_t ws_size,
                              hipStream_t stream) {
    (void)in_sizes; (void)n_in; (void)d_ws; (void)ws_size; (void)out_size;
    const float* x    = (const float*)d_in[0];
    const float* W    = (const float*)d_in[1];
    const float* bias = (const float*)d_in[2];
    float*       out  = (float*)d_out;

    const size_t smem =
        (size_t)(H_TILE * LDS_STRIDE + T_CHUNK * LDS_STRIDE) * sizeof(unsigned short) +
        (size_t)T_CHUNK * CUR_STRIDE * sizeof(float);   // ~158 KB (<= 320 KB/WGP)

    hipFuncSetAttribute((const void*)snn_fused_kernel,
                        hipFuncAttributeMaxDynamicSharedMemorySize, (int)smem);

    dim3 grid(H_DIM / H_TILE, B_DIM);   // (8, 32)
    snn_fused_kernel<<<grid, TPB, smem, stream>>>(x, W, bias, out);
}
